// EncoderBlock_9045201125561
// MI455X (gfx1250) — compile-verified, matrix operand order checked by reading, not run on hardware
//
#include <hip/hip_runtime.h>
#include <math.h>
#include <stdint.h>

#define B_   2
#define S_   2048
#define D_   1024
#define H_   16
#define HD_  64
#define G_   2
#define E_   8
#define M_   1024
#define NT_  (B_*S_)      // 4096 tokens
#define EPS_ 1e-5f

typedef __bf16 bf16;
typedef __attribute__((ext_vector_type(16))) __bf16 bf16x16;
typedef __attribute__((ext_vector_type(8)))  __bf16 bf16x8;
typedef __attribute__((ext_vector_type(4)))  __bf16 bf16x4;
typedef __attribute__((ext_vector_type(8)))  float  f32x8;
typedef __attribute__((ext_vector_type(4)))  int    i32x4;

#define AS1 __attribute__((address_space(1)))
#define AS3 __attribute__((address_space(3)))

// ---------- gfx1250 async global->LDS staging ----------
#if defined(__HIP_DEVICE_COMPILE__) && __has_builtin(__builtin_amdgcn_global_load_async_to_lds_b128)
#define HAVE_ASYNC_LDS 1
#else
#define HAVE_ASYNC_LDS 0
#endif

__device__ __forceinline__ void async_copy16(const bf16* g, bf16* l) {
#if HAVE_ASYNC_LDS
  AS1 i32x4* gp = (AS1 i32x4*)(uintptr_t)g;      // int->AS1 ptr: global AS uses flat VAs
  AS3 i32x4* lp = (AS3 i32x4*)(AS3 void*)l;      // generic->LDS addrspace cast
  __builtin_amdgcn_global_load_async_to_lds_b128(gp, lp, 0, 0);
#else
  *(bf16x8*)l = *(const bf16x8*)g;
#endif
}

__device__ __forceinline__ void wait_async() {
#if HAVE_ASYNC_LDS
#if __has_builtin(__builtin_amdgcn_s_wait_asynccnt)
  __builtin_amdgcn_s_wait_asynccnt(0);
#else
  asm volatile("s_wait_asynccnt 0" ::: "memory");
#endif
#endif
}

// ---------- WMMA helpers ----------
__device__ __forceinline__ bf16x16 frag16(const bf16* p) {
  bf16x8 lo = *(const bf16x8*)p;        // K-chunk [koff, koff+8)
  bf16x8 hi = *(const bf16x8*)(p + 16); // K-chunk [koff+16, koff+24)
  return __builtin_shufflevector(lo, hi, 0,1,2,3,4,5,6,7,8,9,10,11,12,13,14,15);
}

__device__ __forceinline__ f32x8 wmma_bf16(bf16x16 a, bf16x16 b, f32x8 c) {
  return __builtin_amdgcn_wmma_f32_16x16x32_bf16(false, a, false, b, (short)0, c, false, false);
}

// =====================================================================
// Weight pre-pass: W[K][N] fp32 -> WT[N][K] bf16 (LDS-tiled transpose)
// =====================================================================
__global__ __launch_bounds__(256, 1)
void transpose_cvt(const float* __restrict__ W, bf16* __restrict__ WT, int K, int N)
{
  __shared__ bf16 T[64*72];
  const int k0 = blockIdx.x * 64, n0 = blockIdx.y * 64;
  const int tid = threadIdx.x;
#pragma unroll
  for (int i = 0; i < 16; ++i) {
    int idx = tid + i*256;
    int r = idx >> 6, c = idx & 63;                 // r: k, c: n (coalesced read)
    T[c*72 + r] = (bf16)W[(size_t)(k0 + r)*N + n0 + c];
  }
  __syncthreads();
#pragma unroll
  for (int i = 0; i < 16; ++i) {
    int idx = tid + i*256;
    int r = idx >> 6, c = idx & 63;                 // r: n, c: k (coalesced write)
    WT[(size_t)(n0 + r)*K + k0 + c] = T[r*72 + c];
  }
}

// fp32 -> bf16 bulk convert (4 elems/thread)
__global__ __launch_bounds__(256, 1)
void cvt_bf16(const float* __restrict__ X, bf16* __restrict__ Y)
{
  int i = (blockIdx.x * 256 + threadIdx.x) * 4;
  float4 f = *(const float4*)(X + i);
  bf16x4 t; t[0] = (bf16)f.x; t[1] = (bf16)f.y; t[2] = (bf16)f.z; t[3] = (bf16)f.w;
  *(bf16x4*)(Y + i) = t;
}

// =====================================================================
// 128x128 tile GEMM, all-bf16 operands, async-DMA staged, double-buffered
//   Out = act(A[M,K] * BT[N,K]^T + bias)
// =====================================================================
__device__ __forceinline__ void stage_tile(const bf16* src, int rstride, bf16* dst, int tid)
{
  // 128 rows x 32 cols bf16, dst stride 48 (96B rows, 16B aligned)
#pragma unroll
  for (int i = 0; i < 2; ++i) {
    int idx = tid + i*256;                 // 512 x 16B chunks
    int r = idx >> 2, c8 = (idx & 3) << 3;
    async_copy16(src + (size_t)r*rstride + c8, &dst[r*48 + c8]);
  }
}

template<bool RELU, bool GATED, bool OUT_BF16>
__global__ __launch_bounds__(256, 1)
void gemm128(const bf16* __restrict__ A, const bf16* __restrict__ BT,
             const float* __restrict__ bias, void* __restrict__ Outp,
             const float* __restrict__ gate, int gstride, float gscale,
             int Nc, int Kc)
{
  __shared__ bf16 As[2][128*48];
  __shared__ bf16 Bs[2][128*48];

  const int tid  = threadIdx.x;
  const int m0   = blockIdx.y * 128;
  const int n0   = blockIdx.x * 128;
  const int w    = tid >> 5, lane = tid & 31;
  const int wm   = w >> 2,   wn   = w & 3;      // 2x4 wave grid
  const int lrow = lane & 15;
  const int koff = (lane >> 4) << 3;            // 0 or 8

  const bf16* Arow = A  + (size_t)m0 * Kc;
  const bf16* Brow = BT + (size_t)n0 * Kc;

  f32x8 acc[4][2];
#pragma unroll
  for (int mt = 0; mt < 4; ++mt)
#pragma unroll
    for (int nt = 0; nt < 2; ++nt)
#pragma unroll
      for (int j = 0; j < 8; ++j) acc[mt][nt][j] = 0.0f;

  // prologue: stage step 0
  stage_tile(Arow, Kc, As[0], tid);
  stage_tile(Brow, Kc, Bs[0], tid);
  wait_async();
  __syncthreads();

  const int nsteps = Kc >> 5;
  for (int s = 0; s < nsteps; ++s) {
    const int cur = s & 1;
    if (s + 1 < nsteps) {                       // async prefetch next K-slab
      stage_tile(Arow + (s + 1)*32, Kc, As[cur ^ 1], tid);
      stage_tile(Brow + (s + 1)*32, Kc, Bs[cur ^ 1], tid);
    }
    // compute on current buffers (overlaps with in-flight async copies)
    {
      const bf16* as = As[cur];
      const bf16* bs = Bs[cur];
      bf16x16 af[4], bfr[2];
#pragma unroll
      for (int mt = 0; mt < 4; ++mt)
        af[mt] = frag16(&as[(wm*64 + mt*16 + lrow)*48 + koff]);
#pragma unroll
      for (int nt = 0; nt < 2; ++nt)
        bfr[nt] = frag16(&bs[(wn*32 + nt*16 + lrow)*48 + koff]);
#pragma unroll
      for (int mt = 0; mt < 4; ++mt)
#pragma unroll
        for (int nt = 0; nt < 2; ++nt)
          acc[mt][nt] = wmma_bf16(af[mt], bfr[nt], acc[mt][nt]);
    }
    wait_async();
    __syncthreads();
  }

  // ---- epilogue ----
#pragma unroll
  for (int mt = 0; mt < 4; ++mt)
#pragma unroll
    for (int nt = 0; nt < 2; ++nt)
#pragma unroll
      for (int j = 0; j < 8; ++j) {
        int r = m0 + wm*64 + mt*16 + ((lane < 16) ? j : j + 8);
        int c = n0 + wn*32 + nt*16 + lrow;
        float v = acc[mt][nt][j] + bias[c];
        if (RELU) v = fmaxf(v, 0.0f);
        if (GATED) {
          float gw = gate[(size_t)r * gstride] * gscale;
          float* o = (float*)Outp + (size_t)r*Nc + c;
          *o += gw * v;                        // launches serialized on stream
        } else if (OUT_BF16) {
          ((bf16*)Outp)[(size_t)r*Nc + c] = (bf16)v;
        } else {
          ((float*)Outp)[(size_t)r*Nc + c] = v;
        }
      }
}

// =====================================================================
// Flash attention: one block per (b, h, 128-row q tile). Causal.
// =====================================================================
__global__ __launch_bounds__(256, 1)
void flash_attn(const bf16* __restrict__ Q, const bf16* __restrict__ K,
                const bf16* __restrict__ V, bf16* __restrict__ O)
{
  __shared__ bf16 Qs[128*72];     // [r][d]   (144B rows)
  __shared__ bf16 Ks[128*72];     // [sk][d]  == B^T layout [n=sk][k=d]
  __shared__ bf16 VsT[64*136];    // [d][sk]  == B^T layout for P*V
  __shared__ float Srow[128*132]; // scores fp32, 528B rows
  __shared__ float mrow[128], lsum[128], arow[128];
  bf16* Ps = (bf16*)Srow;         // P bf16 aliased over S rows, stride 264

  const int tid  = threadIdx.x;
  const int qt   = blockIdx.x, h = blockIdx.y, b = blockIdx.z;
  const int w    = tid >> 5, lane = tid & 31;
  const int lrow = lane & 15;
  const int koff = (lane >> 4) << 3;
  const size_t tokQ = (size_t)b*S_ + (size_t)qt*128;
  const int hoff = h * HD_;

  // stage Q once
#pragma unroll
  for (int i = 0; i < 4; ++i) {
    int idx = tid + i*256;
    int r = idx >> 3, c8 = (idx & 7) << 3;
    async_copy16(Q + (tokQ + r)*D_ + hoff + c8, &Qs[r*72 + c8]);
  }
  if (tid < 128) { mrow[tid] = -3.0e38f; lsum[tid] = 0.0f; }
  f32x8 oacc[4];
#pragma unroll
  for (int nt = 0; nt < 4; ++nt)
#pragma unroll
    for (int j = 0; j < 8; ++j) oacc[nt][j] = 0.0f;
  wait_async();
  __syncthreads();

  for (int jb = 0; jb <= qt; ++jb) {
    const size_t tokK = (size_t)b*S_ + (size_t)jb*128;
    // stage K (row major, async) + V (transposed)
#pragma unroll
    for (int i = 0; i < 4; ++i) {
      int idx = tid + i*256;
      int r = idx >> 3, c8 = (idx & 7) << 3;
      async_copy16(K + (tokK + r)*D_ + hoff + c8, &Ks[r*72 + c8]);
    }
#pragma unroll
    for (int i = 0; i < 32; ++i) {
      int idx = tid + i*256;
      int sk = idx >> 6, d = idx & 63;
      VsT[d*136 + sk] = V[(tokK + sk)*D_ + hoff + d];
    }
    wait_async();
    __syncthreads();

    // S = (Q K^T) * 1/sqrt(64), wave grid 2x4 over 128x128
    {
      const int wm = w >> 2, wn = w & 3;
      f32x8 sacc[4][2];
#pragma unroll
      for (int mt = 0; mt < 4; ++mt)
#pragma unroll
        for (int nt = 0; nt < 2; ++nt)
#pragma unroll
          for (int j = 0; j < 8; ++j) sacc[mt][nt][j] = 0.0f;
#pragma unroll
      for (int kk = 0; kk < 64; kk += 32) {
        bf16x16 af[4];
#pragma unroll
        for (int mt = 0; mt < 4; ++mt)
          af[mt] = frag16(&Qs[(wm*64 + mt*16 + lrow)*72 + kk + koff]);
#pragma unroll
        for (int nt = 0; nt < 2; ++nt) {
          bf16x16 bb = frag16(&Ks[(wn*32 + nt*16 + lrow)*72 + kk + koff]);
#pragma unroll
          for (int mt = 0; mt < 4; ++mt)
            sacc[mt][nt] = wmma_bf16(af[mt], bb, sacc[mt][nt]);
        }
      }
#pragma unroll
      for (int mt = 0; mt < 4; ++mt)
#pragma unroll
        for (int nt = 0; nt < 2; ++nt)
#pragma unroll
          for (int j = 0; j < 8; ++j) {
            int r = wm*64 + mt*16 + ((lane < 16) ? j : j + 8);
            int c = wn*32 + nt*16 + lrow;
            Srow[r*132 + c] = sacc[mt][nt][j] * 0.125f;
          }
    }
    __syncthreads();

    // online softmax per row (one thread per row); writes P (bf16, aliased)
    if (tid < 128) {
      const int r = tid;
      const int nlim = (jb == qt) ? (r + 1) : 128;   // causal
      float mold = mrow[r], mx = mold;
      for (int n = 0; n < nlim; ++n) mx = fmaxf(mx, Srow[r*132 + n]);
      float alpha = __expf(mold - mx);
      float l = lsum[r] * alpha;
      for (int n = 0; n < 128; ++n) {
        float e = (n < nlim) ? __expf(Srow[r*132 + n] - mx) : 0.0f;
        l += e;
        Ps[r*264 + n] = (bf16)e;   // in-place downconvert: 2n <= 4n, safe
      }
      mrow[r] = mx; lsum[r] = l; arow[r] = alpha;
    }
    __syncthreads();

    // O = O*alpha + P V   (wave w owns rows [w*16, w*16+16))
#pragma unroll
    for (int nt = 0; nt < 4; ++nt)
#pragma unroll
      for (int j = 0; j < 8; ++j) {
        int r = w*16 + ((lane < 16) ? j : j + 8);
        oacc[nt][j] *= arow[r];
      }
#pragma unroll
    for (int kk = 0; kk < 128; kk += 32) {
      bf16x16 af = frag16(&Ps[(w*16 + lrow)*264 + kk + koff]);
#pragma unroll
      for (int nt = 0; nt < 4; ++nt) {
        bf16x16 bb = frag16(&VsT[(nt*16 + lrow)*136 + kk + koff]);
        oacc[nt] = wmma_bf16(af, bb, oacc[nt]);
      }
    }
    __syncthreads();
  }

  // write O / l
#pragma unroll
  for (int nt = 0; nt < 4; ++nt)
#pragma unroll
    for (int j = 0; j < 8; ++j) {
      int r = w*16 + ((lane < 16) ? j : j + 8);
      float val = oacc[nt][j] / lsum[r];
      O[(tokQ + r)*D_ + hoff + nt*16 + lrow] = (bf16)val;
    }
}

// =====================================================================
// add + LayerNorm (one block per token), optional bf16 mirror output
// =====================================================================
__global__ __launch_bounds__(256, 1)
void add_ln(const float* __restrict__ X, const float* __restrict__ Y,
            const float* __restrict__ gam, const float* __restrict__ bet,
            float* __restrict__ outF, bf16* __restrict__ outB)
{
  __shared__ float rs[256], rs2[256];
  const int t = blockIdx.x, tid = threadIdx.x;
  float v[4], s = 0.0f, s2 = 0.0f;
#pragma unroll
  for (int i = 0; i < 4; ++i) {
    int c = tid + i*256;
    float x = X[(size_t)t*D_ + c] + Y[(size_t)t*D_ + c];
    v[i] = x; s += x; s2 += x*x;
  }
  rs[tid] = s; rs2[tid] = s2;
  __syncthreads();
  for (int o = 128; o > 0; o >>= 1) {
    if (tid < o) { rs[tid] += rs[tid + o]; rs2[tid] += rs2[tid + o]; }
    __syncthreads();
  }
  float mu   = rs[0] * (1.0f / D_);
  float var  = rs2[0] * (1.0f / D_) - mu*mu;
  float rstd = rsqrtf(var + EPS_);
#pragma unroll
  for (int i = 0; i < 4; ++i) {
    int c = tid + i*256;
    float y = (v[i] - mu) * rstd * gam[c] + bet[c];
    outF[(size_t)t*D_ + c] = y;
    if (outB) outB[(size_t)t*D_ + c] = (bf16)y;
  }
}

// =====================================================================
// MoE gating: logits (1 thread per (token, g*E+e)) + per-gate softmax
// =====================================================================
__global__ void gate_logits(const float* __restrict__ Xn, const float* __restrict__ gW,
                            const float* __restrict__ gB, float* __restrict__ logits)
{
  int gid = blockIdx.x * blockDim.x + threadIdx.x;    // NT_*16 threads
  int t = gid >> 4, ge = gid & 15;
  int g = ge >> 3, e = ge & 7;
  const float* x = Xn + (size_t)t * D_;
  const float* wc = gW + (size_t)g * D_ * E_ + e;     // gate_W[g][d][e]
  float acc = gB[ge];
  for (int d = 0; d < D_; ++d) acc += x[d] * wc[(size_t)d * E_];
  logits[(size_t)t*16 + ge] = acc;
}

__global__ void gate_softmax(const float* __restrict__ logits, float* __restrict__ gates)
{
  int gid = blockIdx.x * blockDim.x + threadIdx.x;    // NT_*G_ threads
  int t = gid >> 1, g = gid & 1;
  const float* l = logits + (size_t)t*16 + g*8;
  float mx = l[0];
#pragma unroll
  for (int i = 1; i < 8; ++i) mx = fmaxf(mx, l[i]);
  float e[8], s = 0.0f;
#pragma unroll
  for (int i = 0; i < 8; ++i) { e[i] = __expf(l[i] - mx); s += e[i]; }
  float inv = 1.0f / s;
#pragma unroll
  for (int i = 0; i < 8; ++i) gates[(size_t)t*16 + g*8 + i] = e[i] * inv;
}

// =====================================================================
// Host orchestration
// =====================================================================
extern "C" void kernel_launch(void* const* d_in, const int* in_sizes, int n_in,
                              void* d_out, int out_size, void* d_ws, size_t ws_size,
                              hipStream_t stream)
{
  const float* input = (const float*)d_in[0];
  const float* Wq  = (const float*)d_in[1];
  const float* bq  = (const float*)d_in[2];
  const float* Wk  = (const float*)d_in[3];
  const float* bk  = (const float*)d_in[4];
  const float* Wv  = (const float*)d_in[5];
  const float* bv  = (const float*)d_in[6];
  const float* Wo  = (const float*)d_in[7];
  const float* bo  = (const float*)d_in[8];
  const float* ln1g = (const float*)d_in[9];
  const float* ln1b = (const float*)d_in[10];
  const float* gW  = (const float*)d_in[11];
  const float* gB  = (const float*)d_in[12];
  const float* eW1 = (const float*)d_in[13];
  const float* eb1 = (const float*)d_in[14];
  const float* eW2 = (const float*)d_in[15];
  const float* eb2 = (const float*)d_in[16];
  const float* ln2g = (const float*)d_in[17];
  const float* ln2b = (const float*)d_in[18];
  float* out = (float*)d_out;

  char* wsb = (char*)d_ws; size_t off = 0;
  auto alloc = [&](size_t bytes) -> void* {
    void* p = wsb + off; off = (off + bytes + 255) & ~(size_t)255; return p;
  };
  bf16*  inbf   = (bf16*) alloc((size_t)NT_*D_*2);
  bf16*  qb     = (bf16*) alloc((size_t)NT_*D_*2);
  bf16*  kb     = (bf16*) alloc((size_t)NT_*D_*2);
  bf16*  vb     = (bf16*) alloc((size_t)NT_*D_*2);
  bf16*  attnb  = (bf16*) alloc((size_t)NT_*D_*2);
  bf16*  normb  = (bf16*) alloc((size_t)NT_*D_*2);
  bf16*  hidb   = (bf16*) alloc((size_t)NT_*M_*2);
  bf16*  wT     = (bf16*) alloc((size_t)D_*M_*2);     // reused per-weight [N][K] bf16
  float* attnp  = (float*)alloc((size_t)NT_*D_*4);
  float* normf  = (float*)alloc((size_t)NT_*D_*4);
  float* moeacc = (float*)alloc((size_t)NT_*D_*4);
  float* logits = (float*)alloc((size_t)NT_*16*4);
  float* gates  = (float*)alloc((size_t)NT_*16*4);

  (void)hipMemsetAsync(moeacc, 0, (size_t)NT_*D_*4, stream);

  const dim3 gT(D_/64, D_/64);          // 1024x1024 weight transpose
  const dim3 gG(D_/128, NT_/128);       // (8, 32)

  // input -> bf16 once
  cvt_bf16<<<(NT_*D_)/1024, 256, 0, stream>>>(input, inbf);

  // QKV projections (pre-transposed bf16 weights, async/WMMA GEMM)
  transpose_cvt<<<gT,256,0,stream>>>(Wq, wT, D_, D_);
  gemm128<false,false,true><<<gG,256,0,stream>>>(inbf, wT, bq, qb, nullptr,0,0.f, D_, D_);
  transpose_cvt<<<gT,256,0,stream>>>(Wk, wT, D_, D_);
  gemm128<false,false,true><<<gG,256,0,stream>>>(inbf, wT, bk, kb, nullptr,0,0.f, D_, D_);
  transpose_cvt<<<gT,256,0,stream>>>(Wv, wT, D_, D_);
  gemm128<false,false,true><<<gG,256,0,stream>>>(inbf, wT, bv, vb, nullptr,0,0.f, D_, D_);

  // attention
  flash_attn<<<dim3(S_/128, H_, B_),256,0,stream>>>(qb, kb, vb, attnb);

  // output projection (fp32 out)
  transpose_cvt<<<gT,256,0,stream>>>(Wo, wT, D_, D_);
  gemm128<false,false,false><<<gG,256,0,stream>>>(attnb, wT, bo, attnp, nullptr,0,0.f, D_, D_);

  // residual + LN1 -> normed (fp32 + bf16)
  add_ln<<<NT_,256,0,stream>>>(input, attnp, ln1g, ln1b, normf, normb);

  // gates
  gate_logits<<<(NT_*16)/256,256,0,stream>>>(normf, gW, gB, logits);
  gate_softmax<<<(NT_*G_)/256,256,0,stream>>>(logits, gates);

  // dense MoE: 16 experts, gate-weighted accumulate (scale folds the /G)
  for (int g = 0; g < G_; ++g)
    for (int e = 0; e < E_; ++e) {
      int ge = g*E_ + e;
      const float* w1 = eW1 + (size_t)ge * D_ * M_;
      const float* b1 = eb1 + (size_t)ge * M_;
      const float* w2 = eW2 + (size_t)ge * M_ * D_;
      const float* b2 = eb2 + (size_t)ge * D_;
      transpose_cvt<<<gT,256,0,stream>>>(w1, wT, D_, M_);
      gemm128<true,false,true><<<dim3(M_/128, NT_/128),256,0,stream>>>(
          normb, wT, b1, hidb, nullptr,0,0.f, M_, D_);
      transpose_cvt<<<gT,256,0,stream>>>(w2, wT, M_, D_);
      gemm128<false,true,false><<<dim3(D_/128, NT_/128),256,0,stream>>>(
          hidb, wT, b2, moeacc, gates + ge, G_*E_, 1.0f/G_, D_, M_);
    }

  // residual + LN2 -> output
  add_ln<<<NT_,256,0,stream>>>(normf, moeacc, ln2g, ln2b, out, nullptr);

  (void)in_sizes; (void)n_in; (void)out_size; (void)ws_size;
}
